// SLSTM_9019431322085
// MI455X (gfx1250) — compile-verified, matrix operand order checked by reading, not run on hardware
//
#include <hip/hip_runtime.h>
#include <hip/hip_bf16.h>

// Model dims (fixed by reference)
#define Bb   32
#define Ss   512
#define Hh   256
#define Mrows (Bb*Ss)          // 16384
#define Ngate (7*Hh)           // 1792
#define Kcat  (4*Hh)           // 1024

typedef __attribute__((ext_vector_type(16))) _Float16 v16h;
typedef __attribute__((ext_vector_type(8)))  float    v8f;

union Frag { v16h h; uint4 u[2]; };

// ---------------------------------------------------------------------------
// WMMA GEMM:  C[M x N] = A[M x K] * B[N x K]^T + bias[(m>>9)][n]
// A, B row-major f16 (B is N-major: row n = output feature, contiguous K).
// f32 accumulate via v_wmma_f32_16x16x32_f16.
// Block tile 128x128, BK=32, 8 waves (wave32) in a 4(M) x 2(N) grid;
// wave tile 32x64 -> 8 WMMAs per K-step from 6 LDS fragments (1.5 ds/wmma).
// LDS rows padded to 40 halfs so per-lane ds_load_b128 is bank-conflict free.
// ---------------------------------------------------------------------------
#define BM 128
#define BN 128
#define BK 32
#define LDSS 40   // padded row stride in halfs

__global__ __launch_bounds__(256) void gemm_wmma_f16(
    const _Float16* __restrict__ A, int lda,   // halfs
    const _Float16* __restrict__ Bw, int K,    // ldb == K
    const float* __restrict__ bias,            // [Bb x N], row = m>>9
    float* __restrict__ Cout, int N)
{
  __shared__ alignas(16) _Float16 sA[2][BM * LDSS];
  __shared__ alignas(16) _Float16 sB[2][BN * LDSS];

  const int t    = threadIdx.x;
  const int lane = t & 31;
  const int w    = t >> 5;
  const int wm   = (w & 3) * 32;    // wave M offset in tile (4 slots)
  const int wn   = (w >> 2) * 64;   // wave N offset in tile (2 slots)
  const int m0   = blockIdx.y * BM;
  const int n0   = blockIdx.x * BN;

  const int rowl    = lane & 15;
  const int halfsel = (lane & 16) ? 8 : 0;   // ISA 16-bit A/B lane K split

  // global->LDS staging coordinates (16B chunks); 128 rows x 4 chunks each
  const int r0 = t >> 2;            // rows 0..63
  const int r1 = r0 + 64;           // rows 64..127
  const int ac = (t & 3) * 8;       // chunk col (halfs)

  v8f acc[2][4] = {};
  const int nK = K / BK;
  uint4 ra0, ra1, rb0, rb1;

  // prologue: stage 0
  ra0 = *(const uint4*)(A + (size_t)(m0 + r0) * lda + ac);
  ra1 = *(const uint4*)(A + (size_t)(m0 + r1) * lda + ac);
  rb0 = *(const uint4*)(Bw + (size_t)(n0 + r0) * K + ac);
  rb1 = *(const uint4*)(Bw + (size_t)(n0 + r1) * K + ac);
  *(uint4*)&sA[0][r0 * LDSS + ac] = ra0;
  *(uint4*)&sA[0][r1 * LDSS + ac] = ra1;
  *(uint4*)&sB[0][r0 * LDSS + ac] = rb0;
  *(uint4*)&sB[0][r1 * LDSS + ac] = rb1;
  __syncthreads();

  for (int ks = 0; ks < nK; ++ks) {
    const int cur = ks & 1;
    if (ks + 1 < nK) {
      const int kb = (ks + 1) * BK;
      ra0 = *(const uint4*)(A + (size_t)(m0 + r0) * lda + kb + ac);
      ra1 = *(const uint4*)(A + (size_t)(m0 + r1) * lda + kb + ac);
      rb0 = *(const uint4*)(Bw + (size_t)(n0 + r0) * K + kb + ac);
      rb1 = *(const uint4*)(Bw + (size_t)(n0 + r1) * K + kb + ac);
    }
    if (ks + 2 < nK) {   // emits global_prefetch_b8
      const int kp = (ks + 2) * BK;
      __builtin_prefetch(A + (size_t)(m0 + r0) * lda + kp + ac, 0, 1);
      __builtin_prefetch(Bw + (size_t)(n0 + r0) * K + kp + ac, 0, 1);
    }

    Frag a[2], b[4];
#pragma unroll
    for (int mi = 0; mi < 2; ++mi) {
      const int r = wm + mi * 16 + rowl;
      a[mi].u[0] = *(const uint4*)&sA[cur][r * LDSS + halfsel];
      a[mi].u[1] = *(const uint4*)&sA[cur][r * LDSS + 16 + halfsel];
    }
#pragma unroll
    for (int ni = 0; ni < 4; ++ni) {
      const int r = wn + ni * 16 + rowl;
      b[ni].u[0] = *(const uint4*)&sB[cur][r * LDSS + halfsel];
      b[ni].u[1] = *(const uint4*)&sB[cur][r * LDSS + 16 + halfsel];
    }
#pragma unroll
    for (int mi = 0; mi < 2; ++mi)
#pragma unroll
      for (int ni = 0; ni < 4; ++ni)
        acc[mi][ni] = __builtin_amdgcn_wmma_f32_16x16x32_f16(
            false, a[mi].h, false, b[ni].h, (short)0, acc[mi][ni], false, false);

    __syncthreads();
    if (ks + 1 < nK) {
      const int nb = 1 - cur;
      *(uint4*)&sA[nb][r0 * LDSS + ac] = ra0;
      *(uint4*)&sA[nb][r1 * LDSS + ac] = ra1;
      *(uint4*)&sB[nb][r0 * LDSS + ac] = rb0;
      *(uint4*)&sB[nb][r1 * LDSS + ac] = rb1;
      __syncthreads();
    }
  }

  // epilogue: 16x16 f32 D layout — lane L<16: M=r, N=L; lane>=16: M=r+8
  const int mofs = (lane & 16) ? 8 : 0;
#pragma unroll
  for (int mi = 0; mi < 2; ++mi)
#pragma unroll
    for (int ni = 0; ni < 4; ++ni) {
      const int n = n0 + wn + ni * 16 + rowl;
#pragma unroll
      for (int r = 0; r < 8; ++r) {
        const int m = m0 + wm + mi * 16 + mofs + r;
        Cout[(size_t)m * N + n] = acc[mi][ni][r] + bias[(size_t)(m >> 9) * N + n];
      }
    }
}

// ---------------------------------------------------------------------------
// Small / elementwise kernels
// ---------------------------------------------------------------------------
__global__ void k_packW(const float* __restrict__ Wlr, const float* __restrict__ Wc,
                        const float* __restrict__ Wx, _Float16* __restrict__ out) {
  int idx = blockIdx.x * 256 + threadIdx.x;                 // Ngate*Kcat
  if (idx >= Ngate * Kcat) return;
  int n = idx >> 10, k = idx & 1023;
  float w;
  if (k < 512)      w = Wlr[(size_t)n * 512 + k];
  else if (k < 768) w = Wc[(size_t)n * 256 + (k - 512)];
  else              w = Wx[(size_t)n * 256 + (k - 768)];
  out[idx] = (_Float16)w;
}

__global__ void k_cvt16(const float* __restrict__ in, _Float16* __restrict__ out, int n) {
  int i = blockIdx.x * 256 + threadIdx.x;
  if (i < n) out[i] = (_Float16)in[i];
}

__global__ void k_embed(const int* __restrict__ ids, const float* __restrict__ emb,
                        _Float16* __restrict__ wv) {
  int m = blockIdx.x, h = threadIdx.x;
  int tok = ids[m];
  wv[(size_t)m * Hh + h] = (_Float16)emb[(size_t)tok * Hh + h];
}

__global__ void k_avg(const float* __restrict__ x, float* __restrict__ out) {
  int b = blockIdx.x, h = threadIdx.x;
  float a = 0.f;
  for (int s = 0; s < Ss; ++s) a += x[((size_t)b * Ss + s) * Hh + h];
  out[b * Hh + h] = a * (1.f / (float)Ss);
}

// X row = [hb | ha | hh | wv]  (f16)
__global__ void k_buildX(const float* __restrict__ hh, const _Float16* __restrict__ wv16,
                         _Float16* __restrict__ X) {
  int m = blockIdx.x, h = threadIdx.x;
  int s = m & (Ss - 1);
  float hb = 0.f, ha = 0.f;
  if (s >= 1)      hb += hh[(size_t)(m - 1) * Hh + h];
  if (s >= 2)      hb += hh[(size_t)(m - 2) * Hh + h];
  if (s <= Ss - 2) ha += hh[(size_t)(m + 1) * Hh + h];
  if (s <= Ss - 3) ha += hh[(size_t)(m + 2) * Hh + h];
  _Float16* xr = X + (size_t)m * Kcat;
  xr[h]           = (_Float16)hb;
  xr[Hh + h]      = (_Float16)ha;
  xr[2 * Hh + h]  = (_Float16)hh[(size_t)m * Hh + h];
  xr[3 * Hh + h]  = wv16[(size_t)m * Hh + h];
}

// sentence gates: f_hat_g (softmaxed over h), og_g, fgi per-batch bias
__global__ void k_sgate(const float* __restrict__ sv, const float* __restrict__ avg,
                        const float* __restrict__ Wfgg, const float* __restrict__ Wfgh,
                        const float* __restrict__ Wogg, const float* __restrict__ Wogh,
                        const float* __restrict__ Wfgig, const float* __restrict__ gbias,
                        const float* __restrict__ bfg, const float* __restrict__ bfgi,
                        float* __restrict__ fhg_soft, float* __restrict__ ogg,
                        float* __restrict__ fgib) {
  __shared__ float ssv[Hh], sav[Hh], red[Hh];
  int b = blockIdx.x, h = threadIdx.x;
  ssv[h] = sv[b * Hh + h];
  sav[h] = avg[b * Hh + h];
  __syncthreads();
  float dfg = 0.f, dog = 0.f, dfgi = 0.f;
  for (int k = 0; k < Hh; ++k) {
    float s = ssv[k], a = sav[k];
    dfg  += s * Wfgg[(size_t)h * Hh + k] + a * Wfgh[(size_t)h * Hh + k];
    dog  += s * Wogg[(size_t)h * Hh + k] + a * Wogh[(size_t)h * Hh + k];
    dfgi += s * Wfgig[(size_t)h * Hh + k];
  }
  float fg = 1.f / (1.f + __expf(-(dfg + bfg[h])));
  ogg[b * Hh + h]  = 1.f / (1.f + __expf(-(dog + gbias[5 * Hh + h]))); // ref uses gate_bias[5]
  fgib[b * Hh + h] = dfgi + bfgi[h];
  // softmax of fg over h
  red[h] = fg; __syncthreads();
  for (int st = 128; st > 0; st >>= 1) { if (h < st) red[h] = fmaxf(red[h], red[h + st]); __syncthreads(); }
  float mx = red[0]; __syncthreads();
  float e = __expf(fg - mx);
  red[h] = e; __syncthreads();
  for (int st = 128; st > 0; st >>= 1) { if (h < st) red[h] += red[h + st]; __syncthreads(); }
  fhg_soft[b * Hh + h] = e / red[0];
}

// softmax_h(sigmoid(pre_fgi)) * cs, partial-summed over 64 s per block (deterministic)
__global__ void k_scores(const float* __restrict__ prefgi, const float* __restrict__ cs,
                         float* __restrict__ partial) {
  __shared__ float red[Hh];
  int b = blockIdx.x, c = blockIdx.y, h = threadIdx.x;
  float accv = 0.f;
  for (int s = c * 64; s < c * 64 + 64; ++s) {
    int m = b * Ss + s;
    float v = 1.f / (1.f + __expf(-prefgi[(size_t)m * Hh + h]));
    red[h] = v; __syncthreads();
    for (int st = 128; st > 0; st >>= 1) { if (h < st) red[h] = fmaxf(red[h], red[h + st]); __syncthreads(); }
    float mx = red[0]; __syncthreads();
    float e = __expf(v - mx);
    red[h] = e; __syncthreads();
    for (int st = 128; st > 0; st >>= 1) { if (h < st) red[h] += red[h + st]; __syncthreads(); }
    float sum = red[0]; __syncthreads();
    accv += (e / sum) * cs[(size_t)m * Hh + h];
  }
  partial[((size_t)b * 8 + c) * Hh + h] = accv;
}

__global__ void k_scsfin(const float* __restrict__ partial, const float* __restrict__ fhg,
                         const float* __restrict__ ogg, const float* __restrict__ scs_old,
                         float* __restrict__ sv_new, float* __restrict__ scs_new) {
  int b = blockIdx.x, h = threadIdx.x;
  float sum = 0.f;
  for (int c = 0; c < 8; ++c) sum += partial[((size_t)b * 8 + c) * Hh + h];
  float ns = fhg[b * Hh + h] * scs_old[b * Hh + h] + sum;
  scs_new[b * Hh + h] = ns;
  sv_new[b * Hh + h]  = ogg[b * Hh + h] * tanhf(ns);
}

// Gbias[b][n] = new_sv[b] . W_g[n,:] + gate_bias[n]
__global__ void k_gbias(const float* __restrict__ sv, const float* __restrict__ Wg,
                        const float* __restrict__ gb, float* __restrict__ out) {
  __shared__ float ssv[Hh];
  int b = blockIdx.x, g = blockIdx.y, h = threadIdx.x;
  ssv[h] = sv[b * Hh + h];
  __syncthreads();
  int n = g * Hh + h;
  const float* wr = Wg + (size_t)n * Hh;
  float d = 0.f;
  for (int k = 0; k < Hh; ++k) d += ssv[k] * wr[k];
  out[(size_t)b * Ngate + n] = d + gb[n];
}

// gate nonlinearities, joint softmax over 5H, cell/hidden update
__global__ void k_cell(const float* __restrict__ pre, const float* __restrict__ cs,
                       const float* __restrict__ scs, float* __restrict__ nhh,
                       float* __restrict__ ncs) {
  __shared__ float red[Hh];
  int m = blockIdx.x, h = threadIdx.x;
  int b = m >> 9, s = m & (Ss - 1);
  const float* pr = pre + (size_t)m * Ngate;
  float p[7];
#pragma unroll
  for (int g = 0; g < 7; ++g) p[g] = pr[g * Hh + h];
  float sg[5];
  float lmax = -1e30f;
#pragma unroll
  for (int g = 0; g < 5; ++g) { sg[g] = 1.f / (1.f + __expf(-p[g])); lmax = fmaxf(lmax, sg[g]); }
  red[h] = lmax; __syncthreads();
  for (int st = 128; st > 0; st >>= 1) { if (h < st) red[h] = fmaxf(red[h], red[h + st]); __syncthreads(); }
  float gmax = red[0]; __syncthreads();
  float lsum = 0.f;
#pragma unroll
  for (int g = 0; g < 5; ++g) { sg[g] = __expf(sg[g] - gmax); lsum += sg[g]; }
  red[h] = lsum; __syncthreads();
  for (int st = 128; st > 0; st >>= 1) { if (h < st) red[h] += red[h + st]; __syncthreads(); }
  float inv = 1.f / red[0];
  float it = sg[0] * inv, lt = sg[1] * inv, rt = sg[2] * inv, ft = sg[3] * inv, stg = sg[4] * inv;
  float og = 1.f / (1.f + __expf(-p[5]));
  float ug = tanhf(p[6]);
  float cbv = 0.f, cav = 0.f;
  if (s >= 1)      cbv += cs[(size_t)(m - 1) * Hh + h];
  if (s >= 2)      cbv += cs[(size_t)(m - 2) * Hh + h];
  if (s <= Ss - 2) cav += cs[(size_t)(m + 1) * Hh + h];
  if (s <= Ss - 3) cav += cs[(size_t)(m + 2) * Hh + h];
  float csv  = cs[(size_t)m * Hh + h];
  float scsv = scs[b * Hh + h];
  float nc = lt * cbv + ft * csv + rt * cav + stg * scsv + it * ug;
  ncs[(size_t)m * Hh + h] = nc;
  nhh[(size_t)m * Hh + h] = og * tanhf(nc);
}

// output head: [avg|sv] @ W1^T + b1 -> @ Wout^T + bout -> log_softmax ; + sv copy
__global__ void k_head(const float* __restrict__ avg, const float* __restrict__ sv,
                       const float* __restrict__ W1, const float* __restrict__ b1,
                       const float* __restrict__ Wout, const float* __restrict__ bout,
                       float* __restrict__ out) {
  __shared__ float cat[2 * Hh], conc[Hh], fcv[8];
  int b = blockIdx.x, h = threadIdx.x;
  cat[h]      = avg[b * Hh + h];
  cat[Hh + h] = sv[b * Hh + h];
  __syncthreads();
  const float* wr = W1 + (size_t)h * (2 * Hh);
  float d = 0.f;
  for (int k = 0; k < 2 * Hh; ++k) d += cat[k] * wr[k];
  conc[h] = d + b1[h];
  __syncthreads();
  if (h < 5) {
    const float* wo = Wout + h * Hh;
    float f = 0.f;
    for (int k = 0; k < Hh; ++k) f += conc[k] * wo[k];
    fcv[h] = f + bout[h];
  }
  __syncthreads();
  if (h == 0) {
    float mx = -1e30f;
    for (int c = 0; c < 5; ++c) mx = fmaxf(mx, fcv[c]);
    float sm = 0.f;
    for (int c = 0; c < 5; ++c) sm += __expf(fcv[c] - mx);
    float lse = mx + __logf(sm);
    for (int c = 0; c < 5; ++c) out[b * 5 + c] = fcv[c] - lse;
  }
  out[Bb * 5 + b * Hh + h] = sv[b * Hh + h];
}

// ---------------------------------------------------------------------------
extern "C" void kernel_launch(void* const* d_in, const int* in_sizes, int n_in,
                              void* d_out, int out_size, void* d_ws, size_t ws_size,
                              hipStream_t stream) {
  const int*   ids   = (const int*)d_in[0];
  // d_in[1] = sentences_length: unused by the reference
  const float* hh0   = (const float*)d_in[2];
  const float* cs0   = (const float*)d_in[3];
  const float* emb   = (const float*)d_in[4];
  const float* Wlr   = (const float*)d_in[5];
  const float* Wc    = (const float*)d_in[6];
  const float* Wx    = (const float*)d_in[7];
  const float* Wg    = (const float*)d_in[8];
  const float* gbias = (const float*)d_in[9];
  const float* Wfgg  = (const float*)d_in[10];
  const float* Wfgh  = (const float*)d_in[11];
  const float* Wfgig = (const float*)d_in[12];
  const float* Wfgih = (const float*)d_in[13];
  const float* Wogg  = (const float*)d_in[14];
  const float* Wogh  = (const float*)d_in[15];
  const float* bfg   = (const float*)d_in[16];
  const float* bfgi  = (const float*)d_in[17];
  const float* W1    = (const float*)d_in[18];
  const float* b1    = (const float*)d_in[19];
  const float* Wout  = (const float*)d_in[20];
  const float* bout  = (const float*)d_in[21];
  float* out = (float*)d_out;

  // workspace partition (256B aligned)
  char* base = (char*)d_ws;
  size_t off = 0;
  auto carve = [&](size_t bytes) -> char* {
    char* p = base + off;
    off = (off + bytes + 255) & ~(size_t)255;
    return p;
  };
  _Float16* wv16   = (_Float16*)carve((size_t)Mrows * Hh * 2);
  _Float16* X16    = (_Float16*)carve((size_t)Mrows * Kcat * 2);
  _Float16* Wcat16 = (_Float16*)carve((size_t)Ngate * Kcat * 2);
  _Float16* Wfgi16 = (_Float16*)carve((size_t)Hh * Hh * 2);
  float* pre    = (float*)carve((size_t)Mrows * Ngate * 4);
  float* prefgi = (float*)carve((size_t)Mrows * Hh * 4);
  float* hhA    = (float*)carve((size_t)Mrows * Hh * 4);
  float* hhB    = (float*)carve((size_t)Mrows * Hh * 4);
  float* csA    = (float*)carve((size_t)Mrows * Hh * 4);
  float* csB    = (float*)carve((size_t)Mrows * Hh * 4);
  float* svb[2]  = {(float*)carve(Bb * Hh * 4), (float*)carve(Bb * Hh * 4)};
  float* scsb[2] = {(float*)carve(Bb * Hh * 4), (float*)carve(Bb * Hh * 4)};
  float* avg    = (float*)carve(Bb * Hh * 4);
  float* fhg    = (float*)carve(Bb * Hh * 4);
  float* ogg    = (float*)carve(Bb * Hh * 4);
  float* fgib   = (float*)carve(Bb * Hh * 4);
  float* Gbias  = (float*)carve((size_t)Bb * Ngate * 4);
  float* part   = (float*)carve((size_t)Bb * 8 * Hh * 4);

  // one-time packs (weights are inputs; repacked each call — cheap, deterministic)
  k_packW<<<(Ngate * Kcat + 255) / 256, 256, 0, stream>>>(Wlr, Wc, Wx, Wcat16);
  k_cvt16<<<(Hh * Hh + 255) / 256, 256, 0, stream>>>(Wfgih, Wfgi16, Hh * Hh);
  k_embed<<<Mrows, Hh, 0, stream>>>(ids, emb, wv16);
  k_avg<<<Bb, Hh, 0, stream>>>(hh0, svb[0]);
  k_avg<<<Bb, Hh, 0, stream>>>(cs0, scsb[0]);

  const float* hh = hh0;
  const float* cs = cs0;
  float* hhN[2] = {hhA, hhB};
  float* csN[2] = {csA, csB};
  int svcur = 0;

  for (int layer = 0; layer < 2; ++layer) {
    k_avg<<<Bb, Hh, 0, stream>>>(hh, avg);
    k_sgate<<<Bb, Hh, 0, stream>>>(svb[svcur], avg, Wfgg, Wfgh, Wogg, Wogh, Wfgig,
                                   gbias, bfg, bfgi, fhg, ogg, fgib);
    k_buildX<<<Mrows, Hh, 0, stream>>>(hh, wv16, X16);
    // pre_fgi = hh_f16 @ W_fgi_h^T + fgi_bias  (hh_f16 lives at X cols [2H,3H))
    gemm_wmma_f16<<<dim3(Hh / BN, Mrows / BM), 256, 0, stream>>>(
        X16 + 2 * Hh, Kcat, Wfgi16, Hh, fgib, prefgi, Hh);
    k_scores<<<dim3(Bb, 8), Hh, 0, stream>>>(prefgi, cs, part);
    const float* scs_old = scsb[svcur];
    k_scsfin<<<Bb, Hh, 0, stream>>>(part, fhg, ogg, scs_old, svb[1 - svcur], scsb[1 - svcur]);
    svcur ^= 1;
    k_gbias<<<dim3(Bb, 7), Hh, 0, stream>>>(svb[svcur], Wg, gbias, Gbias);
    // pre = X @ Wcat^T + Gbias[b]   (main GEMM: 16384 x 1792 x 1024)
    gemm_wmma_f16<<<dim3(Ngate / BN, Mrows / BM), 256, 0, stream>>>(
        X16, Kcat, Wcat16, Kcat, Gbias, pre, Ngate);
    k_cell<<<Mrows, Hh, 0, stream>>>(pre, cs, scs_old, hhN[layer], csN[layer]);
    hh = hhN[layer];
    cs = csN[layer];
  }

  k_avg<<<Bb, Hh, 0, stream>>>(hh, avg);
  k_head<<<Bb, Hh, 0, stream>>>(avg, svb[svcur], W1, b1, Wout, bout, out);

  (void)in_sizes; (void)n_in; (void)out_size; (void)ws_size;
}
